// RelationalGraphNeuralNetwork_45973329937222
// MI455X (gfx1250) — matrix operand order repaired and checked by reading, not compiled, not run on hardware
//
#include <hip/hip_runtime.h>
#include <hip/hip_bf16.h>

// ---------------------------------------------------------------------------
// CDNA5 (gfx1250) fused relational-GNN.
//  - GEMMs via v_wmma_f32_16x16x32_f16 (wave32, f32 accum)
//  - Weight tiles staged into LDS with TDM tensor_load_to_lds (+ s_wait_tensorcnt)
//  - Scatter-max via native int atomics (sign-split trick)
// ---------------------------------------------------------------------------

typedef _Float16 v16h __attribute__((ext_vector_type(16)));
typedef _Float16 v8h  __attribute__((ext_vector_type(8)));
typedef float    v8f  __attribute__((ext_vector_type(8)));
typedef float    f32x4 __attribute__((ext_vector_type(4)));
typedef unsigned int u32x4 __attribute__((ext_vector_type(4)));
typedef int      i32x8 __attribute__((ext_vector_type(8)));
typedef int      i32x4 __attribute__((ext_vector_type(4)));

__device__ __forceinline__ v8f wmma_f16(v16h a, v16h b, v8f c) {
  return __builtin_amdgcn_wmma_f32_16x16x32_f16(false, a, false, b, (short)0, c,
                                                false, false);
}

// mish(x) = x * tanh(softplus(x)) = x * (1 - 2/(u^2+1)), u = 1+e^x
// (u -> inf: rcp -> 0 -> x; x -> -inf: u -> 1 -> 0). Two trans ops total.
__device__ __forceinline__ float mishf(float x) {
  float e = __expf(x);
  float u = 1.0f + e;
  float t = 1.0f - 2.0f * __builtin_amdgcn_rcpf(u * u + 1.0f);
  return x * t;
}

__device__ __forceinline__ void atomicMaxF(float* addr, float v) {
  if (v >= 0.0f) {
    atomicMax((int*)addr, __float_as_int(v));
  } else {
    atomicMin((unsigned int*)addr, __float_as_uint(v));
  }
}

// TDM: DMA a 2D f16 tile [n_rows][k_len] (contiguous, row stride = k_len) from
// global into LDS. Descriptor per CDNA5 ISA §8.3/8.4: group0 = count/lds/global
// address/type=2 ("image"); group1 = data_size=2B, tensor_dim0/1 = tile_dim0/1,
// dim0_stride = k_len. Groups 2/3 unused (2D tensor). 6-arg builtin form
// (clang-23 / therock-10.0 headers).
__device__ __forceinline__ void tdm_load_2d(const _Float16* gsrc, void* lds_dst,
                                            int k_len, int n_rows) {
  unsigned long long ga = (unsigned long long)(uintptr_t)gsrc;
  unsigned int lds = (unsigned int)(uintptr_t)lds_dst;  // low 32 bits = LDS offset
  u32x4 g0;
  g0[0] = 1u;                                   // count = 1 valid descriptor
  g0[1] = lds;                                  // lds_addr (bytes)
  g0[2] = (unsigned int)(ga & 0xFFFFFFFFu);     // global_addr[31:0]
  g0[3] = (unsigned int)((ga >> 32) & 0x01FFFFFFu) | (2u << 30);  // [56:32]|type=2
  i32x8 g1;
  g1[0] = (1 << 16);                            // data_size = 1 -> 2 bytes
  g1[1] = (k_len & 0xFFFF) << 16;               // tensor_dim0[15:0]
  g1[2] = (n_rows & 0xFFFF) << 16;              // tensor_dim1[15:0]
  g1[3] = (k_len & 0xFFFF) << 16;               // tile_dim0
  g1[4] = (n_rows & 0xFFFF);                    // tile_dim1 (tile_dim2 = 0)
  g1[5] = k_len;                                // tensor_dim0_stride[31:0]
  g1[6] = 0;
  g1[7] = 0;
  i32x4 z4 = {0, 0, 0, 0};
  i32x8 z8 = {0, 0, 0, 0, 0, 0, 0, 0};
  __builtin_amdgcn_tensor_load_to_lds(g0, g1, z4, z4, z8, 0);
}

__global__ __launch_bounds__(256) void init_neg_inf(unsigned int* p, int n) {
  int i = blockIdx.x * blockDim.x + threadIdx.x;
  if (i < n) p[i] = 0xFF800000u;  // -inf
}

// Convert W [K][Nn] f32 row-major -> W^T [Nn][K] f16 contiguous.
__global__ __launch_bounds__(256) void prep_wT(const float* __restrict__ src,
                                               _Float16* __restrict__ dst,
                                               int K, int Nn) {
  int i = blockIdx.x * blockDim.x + threadIdx.x;
  if (i < K * Nn) {
    int k = i / Nn, n = i % Nn;
    dst[n * K + k] = (_Float16)src[i];
  }
}

// ---------------------------------------------------------------------------
// Per-relation kernel: 128 threads (4 waves), 64 tuples per block.
// ---------------------------------------------------------------------------
template <int ARITY>
__global__ __launch_bounds__(128) void rel_kernel(
    const float* __restrict__ emb, const int* __restrict__ idx, int T,
    const _Float16* __restrict__ WiT, const float* __restrict__ bi,
    const _Float16* __restrict__ WoT, const float* __restrict__ bo,
    float* __restrict__ maxmsg) {
  constexpr int D  = ARITY * 64;
  constexpr int DX = D + 8;  // padded stride for x/h tiles

  __shared__ __align__(16) _Float16 wlds[D * D];    // W^T [n][k], DMA'd
  __shared__ __align__(16) _Float16 xlds[64 * DX];  // x tile (f16)
  __shared__ __align__(16) _Float16 hlds[64 * DX];  // hidden tile (f16)
  __shared__ int sidx[64 * ARITY];

  const int tid   = threadIdx.x;
  const int lane  = tid & 31;
  const int wv    = tid >> 5;
  const int t0    = blockIdx.x * 64;
  const int nrows = min(64, T - t0);
  const int nlane = lane & 15;
  const int sel   = lane >> 4;
  const int m0    = wv * 16;

  // Kick off weight DMA (wave 0) so it overlaps the gather.
  if (wv == 0) tdm_load_2d(WiT, wlds, D, D);

  // stage tuple indices
  for (int i = tid; i < 64 * ARITY; i += 128) {
    int t   = t0 + i / ARITY;
    sidx[i] = (t < T) ? idx[t * ARITY + (i % ARITY)] : 0;
  }
  __syncthreads();
  // gather x tile: float4 loads (E=64 -> 16B aligned), cvt to f16
  for (int i = tid; i < 64 * (D / 4); i += 128) {
    int r  = i / (D / 4);
    int c4 = (i % (D / 4)) * 4;
    f32x4 v = {0.f, 0.f, 0.f, 0.f};
    if (r < nrows) {
      int node = sidx[r * ARITY + (c4 >> 6)];
      v = *(const f32x4*)(emb + node * 64 + (c4 & 63));
    }
    _Float16* p = xlds + r * DX + c4;
    p[0] = (_Float16)v[0]; p[1] = (_Float16)v[1];
    p[2] = (_Float16)v[2]; p[3] = (_Float16)v[3];
  }
  if (wv == 0) __builtin_amdgcn_s_wait_tensorcnt((short)0);
  __syncthreads();

  // ================= GEMM1: h = mish(x @ Wi + bi) =================
  for (int nt = 0; nt < D / 16; ++nt) {
    const int ncol = nt * 16 + nlane;
    v8f acc;
    {
      float b = bi[ncol];
#pragma unroll
      for (int i = 0; i < 8; ++i) acc[i] = b;
    }
#pragma unroll
    for (int k = 0; k < D; k += 32) {
      union { v16h v; v8h h[2]; } au, bu;
      const _Float16* ap = xlds + (m0 + nlane) * DX + k + sel * 8;
      au.h[0] = *(const v8h*)(ap);
      au.h[1] = *(const v8h*)(ap + 16);
      const _Float16* bp = wlds + ncol * D + k + sel * 8;
      bu.h[0] = *(const v8h*)(bp);
      bu.h[1] = *(const v8h*)(bp + 16);
      acc = wmma_f16(au.v, bu.v, acc);
    }
#pragma unroll
    for (int i = 0; i < 8; ++i) {
      hlds[(m0 + i + 8 * sel) * DX + ncol] = (_Float16)mishf(acc[i]);
    }
  }
  __syncthreads();
  if (wv == 0) {
    tdm_load_2d(WoT, wlds, D, D);  // overwrite weight buffer with Wo^T
    __builtin_amdgcn_s_wait_tensorcnt((short)0);
  }
  __syncthreads();

  // ============ GEMM2: m = x + h @ Wo + bo, then scatter-max ============
  for (int nt = 0; nt < D / 16; ++nt) {
    const int ncol = nt * 16 + nlane;
    v8f acc;
    {
      float b = bo[ncol];
#pragma unroll
      for (int i = 0; i < 8; ++i) acc[i] = b;
    }
#pragma unroll
    for (int k = 0; k < D; k += 32) {
      union { v16h v; v8h h[2]; } au, bu;
      const _Float16* ap = hlds + (m0 + nlane) * DX + k + sel * 8;
      au.h[0] = *(const v8h*)(ap);
      au.h[1] = *(const v8h*)(ap + 16);
      const _Float16* bp = wlds + ncol * D + k + sel * 8;
      bu.h[0] = *(const v8h*)(bp);
      bu.h[1] = *(const v8h*)(bp + 16);
      acc = wmma_f16(au.v, bu.v, acc);
    }
    const int slot = ncol >> 6;
    const int e    = ncol & 63;
#pragma unroll
    for (int i = 0; i < 8; ++i) {
      int r = m0 + i + 8 * sel;
      if (r < nrows) {
        float v  = acc[i] + (float)xlds[r * DX + ncol];  // residual
        int node = sidx[r * ARITY + slot];
        atomicMaxF(&maxmsg[node * 64 + e], v);
      }
    }
  }
}

// ---------------------------------------------------------------------------
// Final update MLP: out = mish(concat[maxmsg, emb] @ uWi + ubi) @ uWo + ubo
// ---------------------------------------------------------------------------
__global__ __launch_bounds__(128) void final_kernel(
    const float* __restrict__ emb, const float* __restrict__ maxmsg,
    const _Float16* __restrict__ uWiT, const float* __restrict__ ubi,
    const _Float16* __restrict__ uWoT, const float* __restrict__ ubo,
    float* __restrict__ out, int N) {
  constexpr int D  = 128;
  constexpr int DO = 64;
  constexpr int DX = D + 8;

  __shared__ __align__(16) _Float16 wlds[D * D];    // uWi^T / uWo^T (DMA'd)
  __shared__ __align__(16) _Float16 xlds[64 * DX];  // concat tile
  __shared__ __align__(16) _Float16 hlds[64 * DX];

  const int tid   = threadIdx.x;
  const int lane  = tid & 31;
  const int wv    = tid >> 5;
  const int n0    = blockIdx.x * 64;
  const int nrows = min(64, N - n0);
  const int nlane = lane & 15;
  const int sel   = lane >> 4;
  const int m0    = wv * 16;

  if (wv == 0) tdm_load_2d(uWiT, wlds, D, D);

  for (int i = tid; i < 64 * (D / 4); i += 128) {
    int r  = i / (D / 4);
    int c4 = (i % (D / 4)) * 4;
    f32x4 v = {0.f, 0.f, 0.f, 0.f};
    if (r < nrows) {
      int node = n0 + r;
      v = (c4 < 64) ? *(const f32x4*)(maxmsg + node * 64 + c4)
                    : *(const f32x4*)(emb + node * 64 + (c4 - 64));
    }
    _Float16* p = xlds + r * DX + c4;
    p[0] = (_Float16)v[0]; p[1] = (_Float16)v[1];
    p[2] = (_Float16)v[2]; p[3] = (_Float16)v[3];
  }
  if (wv == 0) __builtin_amdgcn_s_wait_tensorcnt((short)0);
  __syncthreads();

  // GEMM1: 128 -> 128, mish
  for (int nt = 0; nt < D / 16; ++nt) {
    const int ncol = nt * 16 + nlane;
    v8f acc;
    {
      float b = ubi[ncol];
#pragma unroll
      for (int i = 0; i < 8; ++i) acc[i] = b;
    }
#pragma unroll
    for (int k = 0; k < D; k += 32) {
      union { v16h v; v8h h[2]; } au, bu;
      const _Float16* ap = xlds + (m0 + nlane) * DX + k + sel * 8;
      au.h[0] = *(const v8h*)(ap);
      au.h[1] = *(const v8h*)(ap + 16);
      const _Float16* bp = wlds + ncol * D + k + sel * 8;
      bu.h[0] = *(const v8h*)(bp);
      bu.h[1] = *(const v8h*)(bp + 16);
      acc = wmma_f16(au.v, bu.v, acc);
    }
#pragma unroll
    for (int i = 0; i < 8; ++i) {
      hlds[(m0 + i + 8 * sel) * DX + ncol] = (_Float16)mishf(acc[i]);
    }
  }
  __syncthreads();
  if (wv == 0) {
    tdm_load_2d(uWoT, wlds, D, DO);  // [64 rows][128 k]
    __builtin_amdgcn_s_wait_tensorcnt((short)0);
  }
  __syncthreads();

  // GEMM2: 128 -> 64, + bias, write f32 output
  for (int nt = 0; nt < DO / 16; ++nt) {
    const int ncol = nt * 16 + nlane;
    v8f acc;
    {
      float b = ubo[ncol];
#pragma unroll
      for (int i = 0; i < 8; ++i) acc[i] = b;
    }
#pragma unroll
    for (int k = 0; k < D; k += 32) {
      union { v16h v; v8h h[2]; } au, bu;
      const _Float16* ap = hlds + (m0 + nlane) * DX + k + sel * 8;
      au.h[0] = *(const v8h*)(ap);
      au.h[1] = *(const v8h*)(ap + 16);
      const _Float16* bp = wlds + ncol * D + k + sel * 8;
      bu.h[0] = *(const v8h*)(bp);
      bu.h[1] = *(const v8h*)(bp + 16);
      acc = wmma_f16(au.v, bu.v, acc);
    }
#pragma unroll
    for (int i = 0; i < 8; ++i) {
      int r = m0 + i + 8 * sel;
      if (r < nrows) out[(n0 + r) * 64 + ncol] = acc[i];
    }
  }
}

// ---------------------------------------------------------------------------
extern "C" void kernel_launch(void* const* d_in, const int* in_sizes, int n_in,
                              void* d_out, int out_size, void* d_ws,
                              size_t ws_size, hipStream_t stream) {
  const float* emb  = (const float*)d_in[0];
  const int*   idx0 = (const int*)d_in[1];
  const int*   idx1 = (const int*)d_in[2];
  const int*   idx2 = (const int*)d_in[3];
  const int*   idx3 = (const int*)d_in[4];

  const float* rWi[4] = {(const float*)d_in[5], (const float*)d_in[9],
                         (const float*)d_in[13], (const float*)d_in[17]};
  const float* rbi[4] = {(const float*)d_in[6], (const float*)d_in[10],
                         (const float*)d_in[14], (const float*)d_in[18]};
  const float* rWo[4] = {(const float*)d_in[7], (const float*)d_in[11],
                         (const float*)d_in[15], (const float*)d_in[19]};
  const float* rbo[4] = {(const float*)d_in[8], (const float*)d_in[12],
                         (const float*)d_in[16], (const float*)d_in[20]};
  const float* uWi = (const float*)d_in[21];
  const float* ubi = (const float*)d_in[22];
  const float* uWo = (const float*)d_in[23];
  const float* ubo = (const float*)d_in[24];

  const int N  = in_sizes[0] / 64;
  const int T[4] = {in_sizes[1], in_sizes[2] / 2, in_sizes[3] / 2,
                    in_sizes[4] / 3};
  const int AR[4] = {1, 2, 2, 3};

  // workspace layout: maxmsg [N*64 f32], then f16 transposed weights
  float* maxmsg = (float*)d_ws;
  _Float16* wbase = (_Float16*)((char*)d_ws + (size_t)N * 64 * sizeof(float));
  _Float16* wT[10];
  {
    size_t off = 0;
    for (int rr = 0; rr < 4; ++rr) {
      int d = AR[rr] * 64;
      wT[2 * rr + 0] = wbase + off; off += (size_t)d * d;  // Wi^T
      wT[2 * rr + 1] = wbase + off; off += (size_t)d * d;  // Wo^T
    }
    wT[8] = wbase + off; off += 128 * 128;  // uWi^T
    wT[9] = wbase + off; off += 64 * 128;   // uWo^T [64][128]
  }

  {
    int n = N * 64;
    init_neg_inf<<<(n + 255) / 256, 256, 0, stream>>>((unsigned int*)maxmsg, n);
  }
  // weight conversion (tiny; runs once per launch)
  for (int rr = 0; rr < 4; ++rr) {
    int d = AR[rr] * 64, nel = d * d;
    prep_wT<<<(nel + 255) / 256, 256, 0, stream>>>(rWi[rr], wT[2 * rr + 0], d, d);
    prep_wT<<<(nel + 255) / 256, 256, 0, stream>>>(rWo[rr], wT[2 * rr + 1], d, d);
  }
  prep_wT<<<(128 * 128 + 255) / 256, 256, 0, stream>>>(uWi, wT[8], 128, 128);
  prep_wT<<<(128 * 64 + 255) / 256, 256, 0, stream>>>(uWo, wT[9], 128, 64);

  rel_kernel<1><<<(T[0] + 63) / 64, 128, 0, stream>>>(
      emb, idx0, T[0], wT[0], rbi[0], wT[1], rbo[0], maxmsg);
  rel_kernel<2><<<(T[1] + 63) / 64, 128, 0, stream>>>(
      emb, idx1, T[1], wT[2], rbi[1], wT[3], rbo[1], maxmsg);
  rel_kernel<2><<<(T[2] + 63) / 64, 128, 0, stream>>>(
      emb, idx2, T[2], wT[4], rbi[2], wT[5], rbo[2], maxmsg);
  rel_kernel<3><<<(T[3] + 63) / 64, 128, 0, stream>>>(
      emb, idx3, T[3], wT[6], rbi[3], wT[7], rbo[3], maxmsg);

  final_kernel<<<(N + 63) / 64, 128, 0, stream>>>(emb, maxmsg, wT[8], ubi,
                                                  wT[9], ubo, (float*)d_out, N);
}
